// GraphInference_17600775979611
// MI455X (gfx1250) — compile-verified, matrix operand order checked by reading, not run on hardware
//
#include <hip/hip_runtime.h>
#include <math.h>

typedef __attribute__((ext_vector_type(16))) _Float16 v16h;
typedef __attribute__((ext_vector_type(8)))  _Float16 v8h;
typedef __attribute__((ext_vector_type(8)))  float    v8f;

#define B_   2
#define C_   64
#define N_   784
#define BN_  (B_*N_)      // 1568
#define NT_  49           // N tiles of 16
#define TRI_ (NT_*(NT_+1)/2)   // 1225 triangular tiles
#define BND_ 49           // block size for "same" mask
#define NEGV (-1e30f)

// ---------------- WMMA tile helpers (wave32, 16x16x32 f16 -> f32) ----------------
// A-matrix (16x32, row r = lane&15): lanes 0-15 hold K=0..7 & 16..23, lanes 16-31 hold K=8..15 & 24..31.
// A B-matrix tile of Wt (W transposed, row j = output col) loads with the same pattern.
__device__ __forceinline__ v16h load_tile16(const _Float16* __restrict__ base, int ld, int row0, int k0) {
  int lane = threadIdx.x & 31;
  const _Float16* p = base + (size_t)(row0 + (lane & 15)) * ld + k0 + ((lane >> 4) << 3);
  v8h lo = *(const v8h*)(p);
  v8h hi = *(const v8h*)(p + 16);
  v16h r;
#pragma unroll
  for (int t = 0; t < 8; ++t) { r[t] = lo[t]; r[t + 8] = hi[t]; }
  return r;
}

// C/D layout: VGPR r, lanes 0-15 -> (M=r, N=lane), lanes 16-31 -> (M=r+8, N=lane-16)
__device__ __forceinline__ void store_tile_f32(float* __restrict__ base, int ld, int row0, int col0, v8f c) {
  int lane = threadIdx.x & 31;
  int col  = col0 + (lane & 15);
  int rb   = row0 + ((lane >> 4) << 3);
#pragma unroll
  for (int r = 0; r < 8; ++r) base[(size_t)(rb + r) * ld + col] = c[r];
}

// transposed store: element (row, col) of the tile written to base[col*ld + row]
__device__ __forceinline__ void store_tile_f32_T(float* __restrict__ base, int ld, int row0, int col0, v8f c) {
  int lane = threadIdx.x & 31;
  int col  = col0 + (lane & 15);
  int rb   = row0 + ((lane >> 4) << 3);
#pragma unroll
  for (int r = 0; r < 8; ++r) base[(size_t)col * ld + (rb + r)] = c[r];
}

__device__ __forceinline__ v8f wmma_f16(v16h a, v16h b, v8f acc) {
  return __builtin_amdgcn_wmma_f32_16x16x32_f16(false, a, false, b, (short)0, acc, false, false);
}

// ---------------- 1. normalize: xt = x / max(||x||,1e-12), (B,C,N) -> (B*N, C) f32 + f16 ----------------
__global__ void k_normalize(const float* __restrict__ x, float* __restrict__ xt, _Float16* __restrict__ xth) {
  __shared__ float s[8][33];
  int tx = threadIdx.x, ty = threadIdx.y;
  int n = blockIdx.x * 32 + tx;
  int b = blockIdx.y;
  if (n >= N_) return;
  float vals[8]; float acc = 0.f;
#pragma unroll
  for (int k = 0; k < 8; ++k) {
    int c = ty + k * 8;
    float v = x[(size_t)b * C_ * N_ + (size_t)c * N_ + n];
    vals[k] = v; acc += v * v;
  }
  s[ty][tx] = acc;
  __syncthreads();
  if (ty == 0) {
    float t = 0.f;
#pragma unroll
    for (int q = 0; q < 8; ++q) t += s[q][tx];
    float nr = sqrtf(t);
    s[0][tx] = 1.0f / fmaxf(nr, 1e-12f);
  }
  __syncthreads();
  float inv = s[0][tx];
  size_t row = (size_t)b * N_ + n;
#pragma unroll
  for (int k = 0; k < 8; ++k) {
    int c = ty + k * 8;
    float v = vals[k] * inv;
    xt[row * C_ + c]  = v;
    xth[row * C_ + c] = (_Float16)v;
  }
}

// ---------------- 2. pack weights to transposed f16 ----------------
__global__ void k_pack(const float* __restrict__ Wgcn, const float* __restrict__ Wfu,
                       _Float16* __restrict__ w1t, _Float16* __restrict__ w2t, _Float16* __restrict__ wfut) {
  int idx = blockIdx.x * blockDim.x + threadIdx.x;
  if (idx < 3 * C_ * C_) {
    int si = idx / (C_ * C_); int r = idx % (C_ * C_);
    int j = r / C_, k = r % C_;            // out col j, in row k
    float wa = Wgcn[(size_t)si * 2 * C_ * C_ + (size_t)k * C_ + j];
    float wb = Wgcn[(size_t)si * 2 * C_ * C_ + (size_t)(C_ + k) * C_ + j];
    w1t[idx] = (_Float16)(wa - wb);
    w2t[idx] = (_Float16)wb;
  }
  if (idx < C_ * C_) {
    int j = idx / C_, k = idx % C_;
    wfut[idx] = (_Float16)Wfu[(size_t)k * C_ + j];
  }
}

// ---------------- 3. adj = XT @ XT^T per batch (WMMA, symmetric: triangular tiles + mirror) ----------------
__global__ void k_adj(const _Float16* __restrict__ xth, float* __restrict__ adj) {
  int w = blockIdx.x * blockDim.y + threadIdx.y;
  if (w >= B_ * TRI_) return;
  int b = w / TRI_; int t = w % TRI_;
  // map t -> (ti, tj) with ti <= tj (uniform per wave -> SALU loop)
  int ti = 0, rem = t;
  while (rem >= NT_ - ti) { rem -= NT_ - ti; ++ti; }
  int tj = ti + rem;
  const _Float16* base = xth + (size_t)b * N_ * C_;
  v8f acc = {0.f,0.f,0.f,0.f,0.f,0.f,0.f,0.f};
#pragma unroll
  for (int k0 = 0; k0 < C_; k0 += 32) {
    v16h a  = load_tile16(base, C_, ti * 16, k0);
    v16h bb = load_tile16(base, C_, tj * 16, k0);   // symmetric: B-tile == A-tile pattern
    acc = wmma_f16(a, bb, acc);
  }
  float* adjb = adj + (size_t)b * N_ * N_;
  store_tile_f32(adjb, N_, ti * 16, tj * 16, acc);
  if (ti != tj) store_tile_f32_T(adjb, N_, ti * 16, tj * 16, acc);   // mirror (adj symmetric)
}

// ---------------- 4. u = xt@(Wa-Wb)+b, v = xt@Wb (WMMA; weights staged via async-to-LDS) ----------------
__global__ void k_uv(const _Float16* __restrict__ xth, const _Float16* __restrict__ w1t,
                     const _Float16* __restrict__ w2t, const float* __restrict__ bgcn,
                     float* __restrict__ u, float* __restrict__ v) {
  // 48KB of f16 weights shared by all 8 waves of the block: stage with async DMA to LDS.
  __shared__ _Float16 sW[2 * 3 * C_ * C_];         // [0..3CC): W1t, [3CC..6CC): W2t
  int tid = threadIdx.y * 32 + threadIdx.x;        // 0..255
  unsigned ldsBase = (unsigned)(size_t)(&sW[0]);   // LDS aperture: low 32 bits = LDS byte address
  const int CC = C_ * C_;
#pragma unroll
  for (int it = 0; it < 12; ++it) {                // 3072 x b128 chunks total / 256 threads
    int chunk = it * 256 + tid;
    const _Float16* g = (chunk < 3 * CC / 8) ? (w1t + (size_t)chunk * 8)
                                             : (w2t + (size_t)(chunk - 3 * CC / 8) * 8);
    unsigned loff = ldsBase + (unsigned)chunk * 16u;
    asm volatile("global_load_async_to_lds_b128 %0, %1, off"
                 :: "v"(loff), "v"(g) : "memory");
  }
  asm volatile("s_wait_asynccnt 0" ::: "memory");
  __syncthreads();

  int w = blockIdx.x * blockDim.y + threadIdx.y;
  const int RT = BN_ / 16;                 // 98 row tiles
  if (w >= 3 * RT * 4) return;
  int si = w / (RT * 4); int r = w % (RT * 4);
  int ti = r / 4, tc = r % 4;
  v8f au = {0.f,0.f,0.f,0.f,0.f,0.f,0.f,0.f};
  v8f av = {0.f,0.f,0.f,0.f,0.f,0.f,0.f,0.f};
#pragma unroll
  for (int k0 = 0; k0 < C_; k0 += 32) {
    v16h a  = load_tile16(xth, C_, ti * 16, k0);
    v16h b1 = load_tile16(&sW[(size_t)si * CC], C_, tc * 16, k0);           // ds loads
    v16h b2 = load_tile16(&sW[(size_t)(3 + si) * CC], C_, tc * 16, k0);     // ds loads
    au = wmma_f16(a, b1, au);
    av = wmma_f16(a, b2, av);
  }
  int lane = threadIdx.x & 31;
  float bg = bgcn[si * C_ + tc * 16 + (lane & 15)];
#pragma unroll
  for (int q = 0; q < 8; ++q) au[q] += bg;
  store_tile_f32(u + (size_t)si * BN_ * C_, C_, ti * 16, tc * 16, au);
  store_tile_f32(v + (size_t)si * BN_ * C_, C_, ti * 16, tc * 16, av);
}

// ---------------- 5. masked max-aggregation; ballot-based sparse scan ----------------
// Wave per (b,i). Each lane loads a DIFFERENT adj value (coalesced); a 32-wide chunk with no
// lane above the lowest threshold is skipped with one ballot test. Hits (rare: 4-5 sigma
// events) are broadcast with shfl and processed cooperatively (2 channels per lane).
__global__ void k_agg(const float* __restrict__ adj, const float* __restrict__ u,
                      const float* __restrict__ v, float* __restrict__ agg) {
  int w = blockIdx.x * blockDim.y + threadIdx.y;
  if (w >= BN_) return;
  int b = w / N_, i = w % N_;
  int lane = threadIdx.x & 31;
  int c0 = lane, c1 = lane + 32;
  size_t row = (size_t)b * N_ + i;
  float u0a = u[(0 * BN_ + row) * C_ + c0], u0b = u[(0 * BN_ + row) * C_ + c1];
  float u1a = u[(1 * BN_ + row) * C_ + c0], u1b = u[(1 * BN_ + row) * C_ + c1];
  float u2a = u[(2 * BN_ + row) * C_ + c0], u2b = u[(2 * BN_ + row) * C_ + c1];
  float m0a = NEGV, m0b = NEGV, m1a = NEGV, m1b = NEGV, m2a = NEGV, m2b = NEGV;
  bool f0 = false, f1 = false, f2 = false;
  const float* adjrow = adj + ((size_t)b * N_ + i) * N_;
  int blkI = i / BND_;
  for (int j0 = 0; j0 < N_; j0 += 32) {
    if (j0 + 64 < N_) __builtin_prefetch(adjrow + j0 + 64 + lane, 0, 0);   // global_prefetch_b8
    int jl = j0 + lane;
    float av = (jl < N_) ? adjrow[jl] : 0.f;     // N_=784 is not a multiple of 32; guard zeroes the last 16 lanes
    unsigned hits = (unsigned)__ballot(av > 0.5f);
    while (hits) {
      int s = __ffs(hits) - 1;
      hits &= hits - 1;
      float a = __shfl(av, s, 32);
      int j = j0 + s;
      bool same = ((j / BND_) == blkI);
      size_t jr = (size_t)b * N_ + j;
      if (same) {                                  // scale 0: a>0.5 && same
        float va = v[(0 * BN_ + jr) * C_ + c0], vb = v[(0 * BN_ + jr) * C_ + c1];
        m0a = fmaxf(m0a, a * (u0a + va)); m0b = fmaxf(m0b, a * (u0b + vb)); f0 = true;
      } else if (a > 0.6f) {                       // scale 1: a>0.6 && !same
        float va = v[(1 * BN_ + jr) * C_ + c0], vb = v[(1 * BN_ + jr) * C_ + c1];
        m1a = fmaxf(m1a, a * (u1a + va)); m1b = fmaxf(m1b, a * (u1b + vb)); f1 = true;
      }
      if (a > 0.7f) {                              // scale 2
        float va = v[(2 * BN_ + jr) * C_ + c0], vb = v[(2 * BN_ + jr) * C_ + c1];
        m2a = fmaxf(m2a, a * (u2a + va)); m2b = fmaxf(m2b, a * (u2b + vb)); f2 = true;
      }
    }
  }
  agg[(0 * BN_ + row) * C_ + c0] = f0 ? m0a : 0.f; agg[(0 * BN_ + row) * C_ + c1] = f0 ? m0b : 0.f;
  agg[(1 * BN_ + row) * C_ + c0] = f1 ? m1a : 0.f; agg[(1 * BN_ + row) * C_ + c1] = f1 ? m1b : 0.f;
  agg[(2 * BN_ + row) * C_ + c0] = f2 ? m2a : 0.f; agg[(2 * BN_ + row) * C_ + c1] = f2 ? m2b : 0.f;
}

// ---------------- 6. BN stats per (si, c) over (B,N) ----------------
__global__ void k_stats(const float* __restrict__ agg, float* __restrict__ mu, float* __restrict__ var) {
  __shared__ float s1[256], s2[256];
  int id = blockIdx.x;                     // 0..191 = si*64+c
  int tid = threadIdx.x;
  float a1 = 0.f, a2 = 0.f;
  for (int r = tid; r < BN_; r += 256) {
    float vv = agg[((size_t)id / C_ * BN_ + r) * C_ + (id % C_)];
    a1 += vv; a2 += vv * vv;
  }
  s1[tid] = a1; s2[tid] = a2;
  __syncthreads();
  for (int off = 128; off > 0; off >>= 1) {
    if (tid < off) { s1[tid] += s1[tid + off]; s2[tid] += s2[tid + off]; }
    __syncthreads();
  }
  if (tid == 0) {
    float m = s1[0] / (float)BN_;
    mu[id]  = m;
    var[id] = s2[0] / (float)BN_ - m * m;
  }
}

// ---------------- 7. BN + residual + attention-sigmoid + relu -> y ----------------
__global__ void k_bn_att(const float* __restrict__ agg, const float* __restrict__ mu, const float* __restrict__ var,
                         const float* __restrict__ gamma, const float* __restrict__ beta,
                         const float* __restrict__ xt, const float* __restrict__ Watt, const float* __restrict__ batt,
                         float* __restrict__ y) {
  int w = blockIdx.x * blockDim.y + threadIdx.y;
  if (w >= BN_) return;
  int lane = threadIdx.x & 31;
  int c0 = lane, c1 = lane + 32;
  size_t row = (size_t)w;
  float x0 = xt[row * C_ + c0], x1 = xt[row * C_ + c1];
#pragma unroll
  for (int si = 0; si < 3; ++si) {
    size_t rr = ((size_t)si * BN_ + row) * C_;
    float r0 = rsqrtf(var[si * C_ + c0] + 1e-5f);
    float r1 = rsqrtf(var[si * C_ + c1] + 1e-5f);
    float h0 = (agg[rr + c0] - mu[si * C_ + c0]) * r0 * gamma[si * C_ + c0] + beta[si * C_ + c0] + x0;
    float h1 = (agg[rr + c1] - mu[si * C_ + c1]) * r1 * gamma[si * C_ + c1] + beta[si * C_ + c1] + x1;
    float part = h0 * Watt[si * C_ + c0] + h1 * Watt[si * C_ + c1];
#pragma unroll
    for (int off = 16; off > 0; off >>= 1) part += __shfl_xor(part, off, 32);
    float att = 1.0f / (1.0f + expf(-(part + batt[si])));
    y[rr + c0] = fmaxf(att * h0, 0.f);
    y[rr + c1] = fmaxf(att * h1, 0.f);
  }
}

// ---------------- 8. p = max over n of y, per (si,b,c) ----------------
__global__ void k_pmax(const float* __restrict__ y, float* __restrict__ p) {
  __shared__ float sm[256];
  int id = blockIdx.x;                 // si*(B*C)+b*C+c
  int si = id / (B_ * C_); int r = id % (B_ * C_);
  int b = r / C_, c = r % C_;
  int tid = threadIdx.x;
  float m = NEGV;
  for (int n = tid; n < N_; n += 256)
    m = fmaxf(m, y[((size_t)si * BN_ + (size_t)b * N_ + n) * C_ + c]);
  sm[tid] = m;
  __syncthreads();
  for (int off = 128; off > 0; off >>= 1) {
    if (tid < off) sm[tid] = fmaxf(sm[tid], sm[tid + off]);
    __syncthreads();
  }
  if (tid == 0) p[(size_t)(si * B_ + b) * C_ + c] = sm[0];
}

// ---------------- 9. tiny MLP per (si,b): h1=relu(p@W1+b1); w=sig(h1@W2+b2); al=w@Wl+bl ----------------
__global__ void k_mlp(const float* __restrict__ p, const float* __restrict__ W1, const float* __restrict__ b1,
                      const float* __restrict__ W2, const float* __restrict__ b2,
                      const float* __restrict__ Wl, const float* __restrict__ bl,
                      float* __restrict__ wgate, float* __restrict__ al) {
  __shared__ float sp[C_], sh[C_], red[C_];
  int blk = blockIdx.x;                // si*B + b
  int si = blk / B_, b = blk % B_;
  int c = threadIdx.x;
  sp[c] = p[(size_t)(si * B_ + b) * C_ + c];
  __syncthreads();
  float a = b1[si * C_ + c];
  for (int k = 0; k < C_; ++k) a += sp[k] * W1[(size_t)si * C_ * C_ + (size_t)k * C_ + c];
  sh[c] = fmaxf(a, 0.f);
  __syncthreads();
  float g = b2[si * C_ + c];
  for (int k = 0; k < C_; ++k) g += sh[k] * W2[(size_t)si * C_ * C_ + (size_t)k * C_ + c];
  g = 1.0f / (1.0f + expf(-g));
  wgate[(size_t)(si * B_ + b) * C_ + c] = g;
  red[c] = g * Wl[si * C_ + c];
  __syncthreads();
  for (int off = 32; off > 0; off >>= 1) {
    if (c < off) red[c] += red[c + off];
    __syncthreads();
  }
  if (c == 0) al[b * 3 + si] = red[0] + bl[si];
}

// ---------------- 10. softmax over scales + fuse -> f16 ----------------
__global__ void k_fuse(const float* __restrict__ al, const float* __restrict__ wgate,
                       const float* __restrict__ y, _Float16* __restrict__ fh) {
  int idx = blockIdx.x * blockDim.x + threadIdx.x;
  if (idx >= BN_ * C_) return;
  int c = idx % C_; int row = idx / C_;
  int b = row / N_;
  float a0 = al[b * 3 + 0], a1 = al[b * 3 + 1], a2 = al[b * 3 + 2];
  float mx = fmaxf(a0, fmaxf(a1, a2));
  float e0 = expf(a0 - mx), e1 = expf(a1 - mx), e2 = expf(a2 - mx);
  float inv = 1.0f / (e0 + e1 + e2);
  float f = e0 * inv * wgate[(size_t)(0 * B_ + b) * C_ + c] * y[((size_t)0 * BN_ + row) * C_ + c]
          + e1 * inv * wgate[(size_t)(1 * B_ + b) * C_ + c] * y[((size_t)1 * BN_ + row) * C_ + c]
          + e2 * inv * wgate[(size_t)(2 * B_ + b) * C_ + c] * y[((size_t)2 * BN_ + row) * C_ + c];
  fh[idx] = (_Float16)f;
}

// ---------------- 11. out = (fused @ Wfu + bfu) transposed to (B,C,N)  (WMMA) ----------------
__global__ void k_outgemm(const _Float16* __restrict__ fh, const _Float16* __restrict__ wfut,
                          const float* __restrict__ bfu, float* __restrict__ out) {
  int w = blockIdx.x * blockDim.y + threadIdx.y;
  if (w >= B_ * NT_ * 4) return;
  int b = w / (NT_ * 4); int r = w % (NT_ * 4);
  int ti = r / 4, tc = r % 4;
  v8f acc = {0.f,0.f,0.f,0.f,0.f,0.f,0.f,0.f};
#pragma unroll
  for (int k0 = 0; k0 < C_; k0 += 32) {
    v16h a  = load_tile16(fh + (size_t)b * N_ * C_, C_, ti * 16, k0);
    v16h bm = load_tile16(wfut, C_, tc * 16, k0);
    acc = wmma_f16(a, bm, acc);
  }
  int lane = threadIdx.x & 31;
  int col = tc * 16 + (lane & 15);
  int rb  = ti * 16 + ((lane >> 4) << 3);
  float bf = bfu[col];
#pragma unroll
  for (int q = 0; q < 8; ++q)
    out[(size_t)b * C_ * N_ + (size_t)col * N_ + (rb + q)] = acc[q] + bf;   // transposed store
}

// ---------------- launch ----------------
extern "C" void kernel_launch(void* const* d_in, const int* in_sizes, int n_in,
                              void* d_out, int out_size, void* d_ws, size_t ws_size,
                              hipStream_t stream) {
  const float* x     = (const float*)d_in[0];
  const float* Wgcn  = (const float*)d_in[1];
  const float* bgcn  = (const float*)d_in[2];
  const float* gamma = (const float*)d_in[3];
  const float* beta  = (const float*)d_in[4];
  const float* Watt  = (const float*)d_in[5];
  const float* batt  = (const float*)d_in[6];
  const float* Wm1   = (const float*)d_in[7];
  const float* bm1   = (const float*)d_in[8];
  const float* Wm2   = (const float*)d_in[9];
  const float* bm2   = (const float*)d_in[10];
  const float* Wl    = (const float*)d_in[11];
  const float* bl    = (const float*)d_in[12];
  const float* Wfu   = (const float*)d_in[13];
  const float* bfu   = (const float*)d_in[14];
  float* out = (float*)d_out;

  char* ws = (char*)d_ws;
  size_t off = 0;
  auto alloc = [&](size_t bytes) { size_t o = off; off = (off + bytes + 255) & ~(size_t)255; return o; };
  float*    xt    = (float*)   (ws + alloc((size_t)BN_ * C_ * 4));
  _Float16* xth   = (_Float16*)(ws + alloc((size_t)BN_ * C_ * 2));
  _Float16* w1t   = (_Float16*)(ws + alloc((size_t)3 * C_ * C_ * 2));
  _Float16* w2t   = (_Float16*)(ws + alloc((size_t)3 * C_ * C_ * 2));
  _Float16* wfut  = (_Float16*)(ws + alloc((size_t)C_ * C_ * 2));
  float*    adj   = (float*)   (ws + alloc((size_t)B_ * N_ * N_ * 4));
  float*    u     = (float*)   (ws + alloc((size_t)3 * BN_ * C_ * 4));
  float*    v     = (float*)   (ws + alloc((size_t)3 * BN_ * C_ * 4));
  float*    agg   = (float*)   (ws + alloc((size_t)3 * BN_ * C_ * 4));
  float*    mu    = (float*)   (ws + alloc((size_t)3 * C_ * 4));
  float*    var   = (float*)   (ws + alloc((size_t)3 * C_ * 4));
  float*    yy    = (float*)   (ws + alloc((size_t)3 * BN_ * C_ * 4));
  float*    p     = (float*)   (ws + alloc((size_t)3 * B_ * C_ * 4));
  float*    wgate = (float*)   (ws + alloc((size_t)3 * B_ * C_ * 4));
  float*    al    = (float*)   (ws + alloc((size_t)B_ * 3 * 4));
  _Float16* fh    = (_Float16*)(ws + alloc((size_t)BN_ * C_ * 2));
  (void)ws_size; (void)in_sizes; (void)n_in; (void)out_size;

  dim3 wblk(32, 8);
  k_normalize<<<dim3((N_ + 31) / 32, B_), dim3(32, 8), 0, stream>>>(x, xt, xth);
  k_pack<<<(3 * C_ * C_ + 255) / 256, 256, 0, stream>>>(Wgcn, Wfu, w1t, w2t, wfut);
  k_adj<<<(B_ * TRI_ + 7) / 8, wblk, 0, stream>>>(xth, adj);
  k_uv<<<(3 * (BN_ / 16) * 4 + 7) / 8, wblk, 0, stream>>>(xth, w1t, w2t, bgcn, u, v);
  k_agg<<<(BN_ + 7) / 8, wblk, 0, stream>>>(adj, u, v, agg);
  k_stats<<<3 * C_, 256, 0, stream>>>(agg, mu, var);
  k_bn_att<<<(BN_ + 7) / 8, wblk, 0, stream>>>(agg, mu, var, gamma, beta, xt, Watt, batt, yy);
  k_pmax<<<3 * B_ * C_, 256, 0, stream>>>(yy, p);
  k_mlp<<<3 * B_, C_, 0, stream>>>(p, Wm1, bm1, Wm2, bm2, Wl, bl, wgate, al);
  k_fuse<<<(BN_ * C_ + 255) / 256, 256, 0, stream>>>(al, wgate, yy, fh);
  k_outgemm<<<(B_ * NT_ * 4 + 7) / 8, wblk, 0, stream>>>(fh, wfut, bfu, out);
}
